// GATv2_85693187490078
// MI455X (gfx1250) — matrix operand order repaired and checked by reading, not compiled, not run on hardware
//
#include <hip/hip_runtime.h>
#include <hip/hip_fp16.h>
#include <hip/hip_bf16.h>

typedef __attribute__((ext_vector_type(16))) _Float16 v16h;
typedef __attribute__((ext_vector_type(8)))  float    v8f;

#define F_IN   14
#define H1     4
#define C1     32
#define HC1    128      // H1*C1
#define C2     21
#define C2P    32       // padded
#define SLOPE  0.2f

// ---------------------------------------------------------------------------
// float atomic max via sign-aware integer ordering (init must be -inf bits)
// ---------------------------------------------------------------------------
__device__ __forceinline__ void atomicMaxFloat(float* addr, float val) {
    if (val >= 0.0f) atomicMax((int*)addr, __float_as_int(val));
    else             atomicMin((unsigned int*)addr, __float_as_uint(val));
}

// WMMA fragment K-index maps (ISA 7.12.2, 16-bit operands, wave32)
__device__ __forceinline__ int kmapA(int lane, int e) {   // A: 16x32, M=lane&15
    return e + (e < 8 ? 0 : 8) + (lane < 16 ? 0 : 8);
}
__device__ __forceinline__ int kmapB(int lane, int e) {   // B: 32x16, N=lane&15
    return e + (e < 8 ? 0 : 8) + (lane < 16 ? 0 : 8);
}

// ---------------------------------------------------------------------------
// init: zero aggregation buffers / output, -inf segment-max, 0 denominators
// ---------------------------------------------------------------------------
__global__ void init_kernel(float* __restrict__ hbuf,
                            float* __restrict__ amax1, float* __restrict__ denom1,
                            float* __restrict__ amax2, float* __restrict__ denom2,
                            float* __restrict__ out, int n) {
    size_t i = (size_t)blockIdx.x * blockDim.x + threadIdx.x;
    const float ninf = __int_as_float((int)0xFF800000);
    if (i < (size_t)n * HC1) hbuf[i] = 0.0f;
    if (i < (size_t)n * H1)  { amax1[i] = ninf; denom1[i] = 0.0f; }
    if (i < (size_t)n)       { amax2[i] = ninf; denom2[i] = 0.0f; }
    if (i < (size_t)n * C2)  out[i] = 0.0f;
}

// ---------------------------------------------------------------------------
// Layer-1 GEMMs: xl1 = x@W1l + b1l ; xr1 = x@W1r + b1r   (N x 14) x (14 x 128)
// 256 threads = 8 waves; wave w -> 16-col tile for BOTH weight matrices.
// LDS holds tiles PRE-SWIZZLED into per-lane fragment order: operand load is
// one aligned 32-byte v16h read (2x ds_load_b128), feeding v_wmma directly.
// ---------------------------------------------------------------------------
__global__ void gemm1_kernel(const float* __restrict__ x,
                             const float* __restrict__ Wl, const float* __restrict__ bl,
                             const float* __restrict__ Wr, const float* __restrict__ br,
                             float* __restrict__ xl1, float* __restrict__ xr1, int n) {
    __shared__ v16h sxf[32];         // A fragments (one per lane)
    __shared__ v16h swlf[8 * 32];    // B fragments (Wl) per wave
    __shared__ v16h swrf[8 * 32];    // B fragments (Wr) per wave
    const int t = threadIdx.x;
    const int row0 = blockIdx.x * 16;

    _Float16* sxp  = (_Float16*)sxf;
    _Float16* swlp = (_Float16*)swlf;
    _Float16* swrp = (_Float16*)swrf;

    // A tile, fragment order: i = lane*16 + e
    for (int i = t; i < 32 * 16; i += 256) {
        int lane = i >> 4, e = i & 15;
        int m = lane & 15;
        int k = kmapA(lane, e);
        int rg = row0 + m; if (rg >= n) rg = n - 1;      // clamp, keep EXEC full
        sxp[i] = (k < F_IN) ? (_Float16)x[(size_t)rg * F_IN + k] : (_Float16)0.0f;
    }
    // B tiles, fragment order: i = (wave*32 + lane)*16 + e
    for (int i = t; i < 8 * 32 * 16; i += 256) {
        int w = i >> 9, lane = (i >> 4) & 31, e = i & 15;
        int k = kmapB(lane, e);
        int col = w * 16 + (lane & 15);
        _Float16 vl = (_Float16)0.0f, vr = (_Float16)0.0f;
        if (k < F_IN) {
            vl = (_Float16)Wl[k * HC1 + col];
            vr = (_Float16)Wr[k * HC1 + col];
        }
        swlp[i] = vl;
        swrp[i] = vr;
    }
    __syncthreads();

    const int wave = t >> 5, lane = t & 31;
    const int col0 = wave * 16;
    const int half = (lane < 16) ? 0 : 1;
    const int m    = lane & 15;

    v16h a  = sxf[lane];                 // 2x ds_load_b128
    v16h bL = swlf[wave * 32 + lane];
    v16h bR = swrf[wave * 32 + lane];

    v8f cl = {}, cr = {};
    cl = __builtin_amdgcn_wmma_f32_16x16x32_f16(false, a, false, bL, (short)0, cl, false, false);
    cr = __builtin_amdgcn_wmma_f32_16x16x32_f16(false, a, false, bR, (short)0, cr, false, false);

#pragma unroll
    for (int i = 0; i < 8; ++i) {
        int om = i + half * 8;
        int oc = col0 + m;
        if (row0 + om < n) {
            xl1[(size_t)(row0 + om) * HC1 + oc] = cl[i] + bl[oc];
            xr1[(size_t)(row0 + om) * HC1 + oc] = cr[i] + br[oc];
        }
    }
}

// ---------------------------------------------------------------------------
// Layer-1 edge pass A: alpha[e][h] = att[h].leakyrelu(xl[src]+xr[dst]); seg-max
// one thread per (edge, head); 128-bit row loads; prefetch future index block
// ---------------------------------------------------------------------------
__global__ void edge_alpha1_kernel(const float* __restrict__ xl1, const float* __restrict__ xr1,
                                   const int* __restrict__ ei, const float* __restrict__ att,
                                   float* __restrict__ alpha1, float* __restrict__ amax1,
                                   int E, int Etot) {
    int g = blockIdx.x * blockDim.x + threadIdx.x;
    if (g >= Etot * H1) return;
    int e = g >> 2, hh = g & 3;
    if (e + 4096 < E) {                       // pull upcoming index stream into cache
        __builtin_prefetch(ei + e + 4096, 0, 0);
        __builtin_prefetch(ei + E + e + 4096, 0, 0);
    }
    int src, dst;
    if (e < E) { src = ei[e]; dst = ei[E + e]; } else { src = dst = e - E; }
    const float4* pl = (const float4*)(xl1 + (size_t)src * HC1 + hh * C1);
    const float4* pr = (const float4*)(xr1 + (size_t)dst * HC1 + hh * C1);
    const float4* pa = (const float4*)(att + hh * C1);
    float s = 0.0f;
#pragma unroll
    for (int q = 0; q < C1 / 4; ++q) {
        float4 l = pl[q], r = pr[q], w = pa[q];
        float v0 = l.x + r.x; v0 = (v0 > 0.0f) ? v0 : SLOPE * v0;
        float v1 = l.y + r.y; v1 = (v1 > 0.0f) ? v1 : SLOPE * v1;
        float v2 = l.z + r.z; v2 = (v2 > 0.0f) ? v2 : SLOPE * v2;
        float v3 = l.w + r.w; v3 = (v3 > 0.0f) ? v3 : SLOPE * v3;
        s += w.x * v0 + w.y * v1 + w.z * v2 + w.w * v3;
    }
    alpha1[g] = s;
    atomicMaxFloat(&amax1[(size_t)dst * H1 + hh], s);
}

// ---------------------------------------------------------------------------
// Layer-1 edge pass B: ex = exp(alpha - amax[dst]); seg-sum denom
// ---------------------------------------------------------------------------
__global__ void edge_exp1_kernel(const int* __restrict__ ei, float* __restrict__ alpha1,
                                 const float* __restrict__ amax1, float* __restrict__ denom1,
                                 int E, int Etot) {
    int g = blockIdx.x * blockDim.x + threadIdx.x;
    if (g >= Etot * H1) return;
    int e = g >> 2, hh = g & 3;
    int dst = (e < E) ? ei[E + e] : (e - E);
    float ex = __expf(alpha1[g] - amax1[(size_t)dst * H1 + hh]);
    alpha1[g] = ex;
    atomicAdd(&denom1[(size_t)dst * H1 + hh], ex);
}

// ---------------------------------------------------------------------------
// Layer-1 edge pass C: hbuf[dst] += a * xl1[src]; one thread per (edge, chan)
// 32 consecutive threads cover channel c of one edge -> coalesced 128B bursts;
// scatter targets live in the 192MB L2, so f32 atomics run at L2 rate.
// ---------------------------------------------------------------------------
__global__ void edge_agg1_kernel(const int* __restrict__ ei, const float* __restrict__ xl1,
                                 const float* __restrict__ alpha1, const float* __restrict__ denom1,
                                 float* __restrict__ hbuf, int E, int Etot) {
    int g = blockIdx.x * blockDim.x + threadIdx.x;
    if (g >= Etot * C1) return;
    int e = g >> 5, c = g & 31;
    int src, dst;
    if (e < E) { src = ei[e]; dst = ei[E + e]; } else { src = dst = e - E; }
#pragma unroll
    for (int hh = 0; hh < H1; ++hh) {
        float a = alpha1[(size_t)e * H1 + hh] / denom1[(size_t)dst * H1 + hh];
        atomicAdd(&hbuf[(size_t)dst * HC1 + hh * C1 + c],
                  a * xl1[(size_t)src * HC1 + hh * C1 + c]);
    }
}

// ---------------------------------------------------------------------------
// h = ELU(agg + bias1)
// ---------------------------------------------------------------------------
__global__ void elu_kernel(float* __restrict__ hbuf, const float* __restrict__ bias1, int n) {
    size_t i = (size_t)blockIdx.x * blockDim.x + threadIdx.x;
    if (i >= (size_t)n * HC1) return;
    float v = hbuf[i] + bias1[i & 127];
    hbuf[i] = (v > 0.0f) ? v : (__expf(v) - 1.0f);
}

// ---------------------------------------------------------------------------
// Layer-2 GEMMs: xl2 = h@W2l + b2l ; xr2 = h@W2r + b2r  (N x 128)x(128 x 21->32)
// 128 threads = 4 waves: wave -> (matrix sel, 16-col tile); K=128 -> 4-WMMA
// accumulation chain. All fragments pre-swizzled in LDS (v16h loads).
// ---------------------------------------------------------------------------
__global__ void gemm2_kernel(const float* __restrict__ h,
                             const float* __restrict__ Wl, const float* __restrict__ bl,
                             const float* __restrict__ Wr, const float* __restrict__ br,
                             float* __restrict__ xl2, float* __restrict__ xr2, int n) {
    __shared__ v16h shf[4 * 32];          // A fragments [kc][lane]
    __shared__ v16h swf[2 * 2 * 4 * 32];  // B fragments [sel][colt][kc][lane]
    const int t = threadIdx.x;            // 128 threads
    const int row0 = blockIdx.x * 16;

    _Float16* shp = (_Float16*)shf;
    _Float16* swp = (_Float16*)swf;

    // A tile: i = (kc*32 + lane)*16 + e
    for (int i = t; i < 4 * 32 * 16; i += 128) {
        int kc = i >> 9, lane = (i >> 4) & 31, e = i & 15;
        int m = lane & 15;
        int ka = kc * 32 + kmapA(lane, e);
        int rg = row0 + m; if (rg >= n) rg = n - 1;
        shp[i] = (_Float16)h[(size_t)rg * HC1 + ka];
    }
    // B tiles: i = (((sel*2 + colt)*4 + kc)*32 + lane)*16 + e
    for (int i = t; i < 2 * 2 * 4 * 32 * 16; i += 128) {
        int sel  = i >> 12;
        int colt = (i >> 11) & 1;
        int kc   = (i >> 9) & 3;
        int lane = (i >> 4) & 31;
        int e    = i & 15;
        int k    = kc * 32 + kmapB(lane, e);
        int col  = colt * 16 + (lane & 15);
        const float* W = sel ? Wr : Wl;
        swp[i] = (col < C2) ? (_Float16)W[k * C2 + col] : (_Float16)0.0f;
    }
    __syncthreads();

    const int wave = t >> 5, lane = t & 31;
    const int sel  = wave >> 1;
    const int colt = wave & 1;
    const int col0 = colt * 16;
    const int half = (lane < 16) ? 0 : 1;
    const int m    = lane & 15;

    v8f c = {};
#pragma unroll
    for (int kc = 0; kc < 4; ++kc) {
        v16h a = shf[kc * 32 + lane];
        v16h b = swf[((sel * 2 + colt) * 4 + kc) * 32 + lane];
        c = __builtin_amdgcn_wmma_f32_16x16x32_f16(false, a, false, b, (short)0, c, false, false);
    }

    float* outp = sel ? xr2 : xl2;
    const float* bp = sel ? br : bl;
#pragma unroll
    for (int i = 0; i < 8; ++i) {
        int om = i + half * 8;
        int oc = col0 + m;
        float v = c[i];
        if (oc < C2) v += bp[oc];
        if (row0 + om < n)
            outp[(size_t)(row0 + om) * C2P + oc] = v;
    }
}

// ---------------------------------------------------------------------------
// Layer-2 edge passes (single head, 21 channels padded to 32)
// ---------------------------------------------------------------------------
__global__ void edge_alpha2_kernel(const float* __restrict__ xl2, const float* __restrict__ xr2,
                                   const int* __restrict__ ei, const float* __restrict__ att2,
                                   float* __restrict__ alpha2, float* __restrict__ amax2,
                                   int E, int Etot) {
    int e = blockIdx.x * blockDim.x + threadIdx.x;
    if (e >= Etot) return;
    int src, dst;
    if (e < E) { src = ei[e]; dst = ei[E + e]; } else { src = dst = e - E; }
    const float* pl = xl2 + (size_t)src * C2P;
    const float* pr = xr2 + (size_t)dst * C2P;
    float s = 0.0f;
#pragma unroll
    for (int c = 0; c < C2; ++c) {
        float v = pl[c] + pr[c];
        v = (v > 0.0f) ? v : SLOPE * v;
        s += att2[c] * v;
    }
    alpha2[e] = s;
    atomicMaxFloat(&amax2[dst], s);
}

__global__ void edge_exp2_kernel(const int* __restrict__ ei, float* __restrict__ alpha2,
                                 const float* __restrict__ amax2, float* __restrict__ denom2,
                                 int E, int Etot) {
    int e = blockIdx.x * blockDim.x + threadIdx.x;
    if (e >= Etot) return;
    int dst = (e < E) ? ei[E + e] : (e - E);
    float ex = __expf(alpha2[e] - amax2[dst]);
    alpha2[e] = ex;
    atomicAdd(&denom2[dst], ex);
}

__global__ void edge_agg2_kernel(const int* __restrict__ ei, const float* __restrict__ xl2,
                                 const float* __restrict__ alpha2, const float* __restrict__ denom2,
                                 float* __restrict__ out, int E, int Etot) {
    int g = blockIdx.x * blockDim.x + threadIdx.x;
    if (g >= Etot * C2P) return;
    int e = g >> 5, c = g & 31;
    if (c >= C2) return;
    int src, dst;
    if (e < E) { src = ei[e]; dst = ei[E + e]; } else { src = dst = e - E; }
    float a = alpha2[e] / denom2[dst];
    atomicAdd(&out[(size_t)dst * C2 + c], a * xl2[(size_t)src * C2P + c]);
}

__global__ void bias2_kernel(float* __restrict__ out, const float* __restrict__ bias2, int n) {
    int i = blockIdx.x * blockDim.x + threadIdx.x;
    if (i >= n * C2) return;
    out[i] += bias2[i % C2];
}

// ---------------------------------------------------------------------------
extern "C" void kernel_launch(void* const* d_in, const int* in_sizes, int n_in,
                              void* d_out, int out_size, void* d_ws, size_t ws_size,
                              hipStream_t stream) {
    const float* x     = (const float*)d_in[0];
    const int*   ei    = (const int*)  d_in[1];
    const float* W1l   = (const float*)d_in[2];
    const float* b1l   = (const float*)d_in[3];
    const float* W1r   = (const float*)d_in[4];
    const float* b1r   = (const float*)d_in[5];
    const float* att1  = (const float*)d_in[6];
    const float* bias1 = (const float*)d_in[7];
    const float* W2l   = (const float*)d_in[8];
    const float* b2l   = (const float*)d_in[9];
    const float* W2r   = (const float*)d_in[10];
    const float* b2r   = (const float*)d_in[11];
    const float* att2  = (const float*)d_in[12];
    const float* bias2 = (const float*)d_in[13];
    float* out = (float*)d_out;

    const int n    = in_sizes[0] / F_IN;   // 100000
    const int E    = in_sizes[1] / 2;      // 800000
    const int Etot = E + n;                // + self loops

    // workspace carve-up (floats)
    float* ws = (float*)d_ws;
    size_t off = 0;
    float* xl1    = ws + off; off += (size_t)n * HC1;
    float* xr1    = ws + off; off += (size_t)n * HC1;
    float* hbuf   = ws + off; off += (size_t)n * HC1;
    float* xl2    = ws + off; off += (size_t)n * C2P;
    float* xr2    = ws + off; off += (size_t)n * C2P;
    float* alpha1 = ws + off; off += (size_t)Etot * H1;
    float* alpha2 = ws + off; off += (size_t)Etot;
    float* amax1  = ws + off; off += (size_t)n * H1;
    float* denom1 = ws + off; off += (size_t)n * H1;
    float* amax2  = ws + off; off += (size_t)n;
    float* denom2 = ws + off; off += (size_t)n;

    const int TB = 256;
    // 1. init
    {
        size_t tot = (size_t)n * HC1;
        init_kernel<<<(unsigned)((tot + TB - 1) / TB), TB, 0, stream>>>(
            hbuf, amax1, denom1, amax2, denom2, out, n);
    }
    // 2. layer-1 GEMMs (WMMA)
    gemm1_kernel<<<(n + 15) / 16, 256, 0, stream>>>(x, W1l, b1l, W1r, b1r, xl1, xr1, n);
    // 3-5. layer-1 edge softmax + aggregation
    edge_alpha1_kernel<<<(Etot * H1 + TB - 1) / TB, TB, 0, stream>>>(
        xl1, xr1, ei, att1, alpha1, amax1, E, Etot);
    edge_exp1_kernel<<<(Etot * H1 + TB - 1) / TB, TB, 0, stream>>>(
        ei, alpha1, amax1, denom1, E, Etot);
    edge_agg1_kernel<<<(Etot * C1 + TB - 1) / TB, TB, 0, stream>>>(
        ei, xl1, alpha1, denom1, hbuf, E, Etot);
    // 6. bias + ELU
    {
        size_t tot = (size_t)n * HC1;
        elu_kernel<<<(unsigned)((tot + TB - 1) / TB), TB, 0, stream>>>(hbuf, bias1, n);
    }
    // 7. layer-2 GEMMs (WMMA)
    gemm2_kernel<<<(n + 15) / 16, 128, 0, stream>>>(hbuf, W2l, b2l, W2r, b2r, xl2, xr2, n);
    // 8-10. layer-2 edge softmax + aggregation into d_out
    edge_alpha2_kernel<<<(Etot + TB - 1) / TB, TB, 0, stream>>>(
        xl2, xr2, ei, att2, alpha2, amax2, E, Etot);
    edge_exp2_kernel<<<(Etot + TB - 1) / TB, TB, 0, stream>>>(
        ei, alpha2, amax2, denom2, E, Etot);
    edge_agg2_kernel<<<(Etot * C2P + TB - 1) / TB, TB, 0, stream>>>(
        ei, xl2, alpha2, denom2, out, E, Etot);
    // 11. + bias2
    bias2_kernel<<<(n * C2 + TB - 1) / TB, TB, 0, stream>>>(out, bias2, n);
}
